// CorrNet_64939905516268
// MI455X (gfx1250) — matrix operand order repaired and checked by reading, not compiled, not run on hardware
//
#include <hip/hip_runtime.h>
#include <hip/hip_bf16.h>

typedef _Float16 v16h __attribute__((ext_vector_type(16)));
typedef _Float16 v8h  __attribute__((ext_vector_type(8)));
typedef float    v8f  __attribute__((ext_vector_type(8)));

#define NEG_SLOPE 0.2f
#define EPS_GN    1e-5f
#define NEG_BIG  -3.402823466e38f

__device__ __forceinline__ float wsum32(float v){
#pragma unroll
  for (int off = 16; off > 0; off >>= 1) v += __shfl_xor(v, off, 32);
  return v;
}

__device__ __forceinline__ void atomicMaxF(float* addr, float val){
  int old = __float_as_int(*addr);
  while (__int_as_float(old) < val){
    int assumed = old;
    old = atomicCAS((int*)addr, assumed, __float_as_int(val));
    if (old == assumed) break;
  }
}

// One wave computes y = GroupNorm(LeakyReLU(W x + b)); group size is always 32
// in this net, which maps exactly onto wave32. xin/yout are per-wave LDS slices.
__device__ __forceinline__ void wave_block(const float* xin, int cin,
                                           float* yout, int cout,
                                           const float* W, const float* b,
                                           const float* gamma, const float* beta,
                                           int lane){
  for (int oc = lane; oc < cout; oc += 32){
    const float* wr = W + (size_t)oc * cin;
    float acc = b[oc];
    for (int k = 0; k < cin; ++k) acc = fmaf(xin[k], wr[k], acc);
    yout[oc] = acc >= 0.f ? acc : NEG_SLOPE * acc;
  }
  __syncthreads();
  const int ng = cout >> 5;
  for (int g = 0; g < ng; ++g){
    const int ch = (g << 5) + lane;
    float v  = yout[ch];
    float s  = wsum32(v);
    float s2 = wsum32(v * v);
    float mu  = s  * (1.f / 32.f);
    float var = s2 * (1.f / 32.f) - mu * mu;
    float xn  = (v - mu) * rsqrtf(var + EPS_GN);
    yout[ch] = xn * gamma[ch] + beta[ch];
  }
  __syncthreads();
}

__global__ void __launch_bounds__(128)
k_init(float* gv, float* gp){
  int t = threadIdx.x;
  if (t < 128){ gv[t] = NEG_BIG; gp[t] = NEG_BIG; }
}

// m1_0 -> m1_1 -> m1_2 (fi stored) -> m2 (hi), channelwise global max of hi.
__global__ void __launch_bounds__(256)
k_front(const float* __restrict__ X, int n,
        const float* W0,const float* b0,const float* g0,const float* e0,
        const float* W1,const float* b1,const float* g1,const float* e1,
        const float* W2,const float* b2,const float* g2,const float* e2,
        const float* W3,const float* b3,const float* g3,const float* e3,
        float* __restrict__ fi, float* __restrict__ gmax){
  __shared__ float bufA[8][256];
  __shared__ float bufB[8][256];
  const int lane = threadIdx.x & 31;
  const int wv   = threadIdx.x >> 5;
  float* A = bufA[wv];
  float* B = bufB[wv];
  const int wgid = blockIdx.x * 8 + wv;
  const int tw   = gridDim.x * 8;
  float lm[4];
#pragma unroll
  for (int j = 0; j < 4; ++j) lm[j] = NEG_BIG;
  const int iters = (n + tw - 1) / tw;     // uniform across the grid
  for (int i = 0; i < iters; ++i){
    const int r = wgid + i * tw;
    const bool valid = r < n;
    const int rr = valid ? r : n - 1;
    if (lane < 3) A[lane] = X[(size_t)rr * 3 + lane];
    __syncthreads();
    wave_block(A, 3,   B, 32,  W0, b0, g0, e0, lane);
    wave_block(B, 32,  A, 64,  W1, b1, g1, e1, lane);
    wave_block(A, 64,  B, 128, W2, b2, g2, e2, lane);
    if (valid){
#pragma unroll
      for (int j = 0; j < 4; ++j)
        fi[(size_t)rr * 128 + lane + 32 * j] = B[lane + 32 * j];
    }
    wave_block(B, 128, A, 128, W3, b3, g3, e3, lane);
    if (valid){
#pragma unroll
      for (int j = 0; j < 4; ++j) lm[j] = fmaxf(lm[j], A[lane + 32 * j]);
    }
    __syncthreads();
  }
#pragma unroll
  for (int j = 0; j < 4; ++j) atomicMaxF(gmax + lane + 32 * j, lm[j]);
}

// concat(fi, g) -> m3_0 -> m3_1 -> lin -> l2norm; f32 out + f16 copy for WMMA.
__global__ void __launch_bounds__(256)
k_back(const float* __restrict__ fi, const float* __restrict__ gmax, int n,
       const float* W4,const float* b4,const float* g4,const float* e4,
       const float* W5,const float* b5,const float* g5,const float* e5,
       const float* linW, const float* linb,
       float* __restrict__ outF, _Float16* __restrict__ outH){
  __shared__ float bufA[8][256];
  __shared__ float bufB[8][256];
  const int lane = threadIdx.x & 31;
  const int wv   = threadIdx.x >> 5;
  float* A = bufA[wv];
  float* B = bufB[wv];
  const int r = blockIdx.x * 8 + wv;
  const bool valid = r < n;
  const int rr = valid ? r : n - 1;
#pragma unroll
  for (int j = 0; j < 4; ++j){
    A[lane + 32 * j]       = fi[(size_t)rr * 128 + lane + 32 * j];
    A[128 + lane + 32 * j] = gmax[lane + 32 * j];
  }
  __syncthreads();
  wave_block(A, 256, B, 128, W4, b4, g4, e4, lane);
  wave_block(B, 128, A, 64,  W5, b5, g5, e5, lane);
#pragma unroll
  for (int j = 0; j < 2; ++j){
    const int oc = lane + 32 * j;
    const float* wr = linW + (size_t)oc * 64;
    float acc = linb[oc];
    for (int k = 0; k < 64; ++k) acc = fmaf(A[k], wr[k], acc);
    B[oc] = acc;
  }
  __syncthreads();
  float o0 = B[lane], o1 = B[lane + 32];
  float nrm2 = wsum32(o0 * o0 + o1 * o1);
  float inv = 1.f / fmaxf(sqrtf(nrm2), 1e-12f);
  o0 *= inv; o1 *= inv;
  if (valid){
    outF[(size_t)rr * 64 + lane]      = o0;
    outF[(size_t)rr * 64 + lane + 32] = o1;
    outH[(size_t)rr * 64 + lane]      = (_Float16)o0;
    outH[(size_t)rr * 64 + lane + 32] = (_Float16)o1;
  }
}

// 16-bit A 16x32 tile: lane L holds row L%16; K elems {8h..8h+7} U {16+8h..16+8h+7}.
__device__ __forceinline__ v16h load_a_tile(const _Float16* base, int row0, int lane, int kc){
  const int r = row0 + (lane & 15);
  const int h = (lane >> 4) & 1;
  const _Float16* p = base + ((size_t)r << 6) + (kc << 5) + (h << 3);
  v8h lo = *reinterpret_cast<const v8h*>(p);
  v8h hi = *reinterpret_cast<const v8h*>(p + 16);
  v16h out;
#pragma unroll
  for (int i = 0; i < 8; ++i){ out[i] = lo[i]; out[i + 8] = hi[i]; }
  return out;
}

// 16-bit B 32x16 tile: lane L holds column L%16; K elems {8h..8h+7} U {16+8h..16+8h+7}.
__device__ __forceinline__ v16h load_b_tile(const _Float16* base, int row0, int lane, int kc){
  const int r = row0 + (lane & 15);
  const int h = (lane >> 4) & 1;
  const _Float16* p = base + ((size_t)r << 6) + (kc << 5) + (h << 3);
  v8h lo = *reinterpret_cast<const v8h*>(p);
  v8h hi = *reinterpret_cast<const v8h*>(p + 16);
  v16h out;
#pragma unroll
  for (int i = 0; i < 8; ++i){ out[i] = lo[i]; out[i + 8] = hi[i]; }
  return out;
}

#define STRIPS 4   // 4 x 16 = 64 vtx rows per block; B tiles reused 4x

// Fused WMMA GEMM + row max/argmax. Block = 64 vtx rows (4 strips); 8 waves
// each sweep M/8 pts columns. Per B tile: 8x v_wmma_f32_16x16x32_f16.
__global__ void __launch_bounds__(256)
k_sim(const _Float16* __restrict__ va, const _Float16* __restrict__ vb,
      float* __restrict__ sim, int* __restrict__ sidx, int M){
  __shared__ float cval[16 * STRIPS][8];
  __shared__ int   cidx[16 * STRIPS][8];
  const int lane  = threadIdx.x & 31;
  const int wv    = threadIdx.x >> 5;
  const int row0  = blockIdx.x * (16 * STRIPS);
  v16h a[STRIPS][2];
#pragma unroll
  for (int s = 0; s < STRIPS; ++s){
    a[s][0] = load_a_tile(va, row0 + 16 * s, lane, 0);
    a[s][1] = load_a_tile(va, row0 + 16 * s, lane, 1);
  }
  float best[STRIPS][8]; int bidx[STRIPS][8];
#pragma unroll
  for (int s = 0; s < STRIPS; ++s)
#pragma unroll
    for (int v = 0; v < 8; ++v){ best[s][v] = NEG_BIG; bidx[s][v] = 0; }
  const int chunk = M >> 3;
  const int mlo = wv * chunk, mhi = mlo + chunk;
  for (int p0 = mlo; p0 < mhi; p0 += 16){
    const v16h b0 = load_b_tile(vb, p0, lane, 0);
    const v16h b1 = load_b_tile(vb, p0, lane, 1);
    v8f acc[STRIPS];
#pragma unroll
    for (int s = 0; s < STRIPS; ++s){
      v8f z = {};
      z = __builtin_amdgcn_wmma_f32_16x16x32_f16(false, a[s][0], false, b0, (short)0, z, false, false);
      z = __builtin_amdgcn_wmma_f32_16x16x32_f16(false, a[s][1], false, b1, (short)0, z, false, false);
      acc[s] = z;
    }
    const int col = p0 + (lane & 15);
#pragma unroll
    for (int s = 0; s < STRIPS; ++s){
#pragma unroll
      for (int v = 0; v < 8; ++v){
        const float q = acc[s][v];
        const bool t = q > best[s][v];       // ties keep earliest (ascending p0)
        best[s][v] = t ? q   : best[s][v];
        bidx[s][v] = t ? col : bidx[s][v];
      }
    }
  }
  const int h = lane >> 4;
#pragma unroll
  for (int s = 0; s < STRIPS; ++s){
#pragma unroll
    for (int v = 0; v < 8; ++v){
      float bv = best[s][v]; int bi = bidx[s][v];
#pragma unroll
      for (int off = 1; off < 16; off <<= 1){  // stays inside the 16-lane half
        float ov = __shfl_xor(bv, off, 32);
        int   oi = __shfl_xor(bi, off, 32);
        if (ov > bv || (ov == bv && oi < bi)){ bv = ov; bi = oi; }
      }
      if ((lane & 15) == 0){
        cval[16 * s + v + (h << 3)][wv] = bv;
        cidx[16 * s + v + (h << 3)][wv] = bi;
      }
    }
  }
  __syncthreads();
  if (wv < STRIPS && lane < 16){
    const int row = 16 * wv + lane;          // wave w reduces strip w's 16 rows
    float bv = cval[row][0]; int bi = cidx[row][0];
#pragma unroll
    for (int w = 1; w < 8; ++w){
      float ov = cval[row][w]; int oi = cidx[row][w];
      if (ov > bv || (ov == bv && oi < bi)){ bv = ov; bi = oi; }
    }
    sim [row0 + row] = bv;
    sidx[row0 + row] = bi;
  }
}

// c = [out_vtx, out_pts[idx], exact-f32 sim] -> corr_mlp -> corr linear.
__global__ void __launch_bounds__(256)
k_corr(const float* __restrict__ outv, const float* __restrict__ outp,
       const int* __restrict__ sidx, int n,
       const float* Wc,const float* bc,const float* gc,const float* ec,
       const float* corrW, const float* corrb,
       float* __restrict__ mask){
  __shared__ float bufA[8][132];
  __shared__ float bufB[8][64];
  const int lane = threadIdx.x & 31;
  const int wv   = threadIdx.x >> 5;
  float* A = bufA[wv];
  float* B = bufB[wv];
  const int r = blockIdx.x * 8 + wv;
  const bool valid = r < n;
  const int rr = valid ? r : n - 1;
  const int idx = sidx[rr];
  const float v0 = outv[(size_t)rr * 64 + lane];
  const float v1 = outv[(size_t)rr * 64 + lane + 32];
  const float c0 = outp[(size_t)idx * 64 + lane];
  const float c1 = outp[(size_t)idx * 64 + lane + 32];
  A[lane]      = v0; A[lane + 32] = v1;
  A[64 + lane] = c0; A[96 + lane] = c1;
  const float s = wsum32(v0 * c0 + v1 * c1);   // exact f32 cosine for chosen idx
  if (lane == 0) A[128] = s;
  __syncthreads();
  wave_block(A, 129, B, 64, Wc, bc, gc, ec, lane);
  const float q = wsum32(B[lane] * corrW[lane] + B[lane + 32] * corrW[lane + 32]);
  if (valid && lane == 0) mask[rr] = q + corrb[0];
}

extern "C" void kernel_launch(void* const* d_in, const int* in_sizes, int n_in,
                              void* d_out, int out_size, void* d_ws, size_t ws_size,
                              hipStream_t stream){
  (void)n_in; (void)out_size; (void)ws_size;
  const float* vtx = (const float*)d_in[0];
  const float* pts = (const float*)d_in[1];
  // params flattened in setup_inputs() insertion order; each block = W,b,g,beta
  const float *W10=(const float*)d_in[2],  *b10=(const float*)d_in[3],
              *g10=(const float*)d_in[4],  *e10=(const float*)d_in[5];
  const float *W11=(const float*)d_in[6],  *b11=(const float*)d_in[7],
              *g11=(const float*)d_in[8],  *e11=(const float*)d_in[9];
  const float *W12=(const float*)d_in[10], *b12=(const float*)d_in[11],
              *g12=(const float*)d_in[12], *e12=(const float*)d_in[13];
  const float *W2 =(const float*)d_in[14], *b2 =(const float*)d_in[15],
              *g2 =(const float*)d_in[16], *e2 =(const float*)d_in[17];
  const float *W30=(const float*)d_in[18], *b30=(const float*)d_in[19],
              *g30=(const float*)d_in[20], *e30=(const float*)d_in[21];
  const float *W31=(const float*)d_in[22], *b31=(const float*)d_in[23],
              *g31=(const float*)d_in[24], *e31=(const float*)d_in[25];
  const float *linW=(const float*)d_in[26], *linb=(const float*)d_in[27];
  const float *Wc =(const float*)d_in[28], *bc =(const float*)d_in[29],
              *gc =(const float*)d_in[30], *ec =(const float*)d_in[31];
  const float *corrW=(const float*)d_in[32], *corrb=(const float*)d_in[33];

  const int n = in_sizes[0] / 3;   // 16384
  const int m = in_sizes[1] / 3;   // 16384
  const size_t NN = (size_t)n, MM = (size_t)m;

  char* ws = (char*)d_ws;
  size_t off = 0;
  auto take = [&](size_t bytes)->void*{
    void* p = ws + off;
    off += (bytes + 255) & ~(size_t)255;
    return p;
  };
  float*    fiV  = (float*)take(NN * 128 * 4);
  float*    fiP  = (float*)take(MM * 128 * 4);
  float*    gV   = (float*)take(128 * 4);
  float*    gP   = (float*)take(128 * 4);
  _Float16* hV   = (_Float16*)take(NN * 64 * 2);
  _Float16* hP   = (_Float16*)take(MM * 64 * 2);
  float*    simv = (float*)take(NN * 4);
  int*      idxv = (int*)take(NN * 4);

  float* outV    = (float*)d_out;
  float* outP    = outV + NN * 64;
  float* outMask = outP + MM * 64;

  k_init<<<1, 128, 0, stream>>>(gV, gP);
  k_front<<<256, 256, 0, stream>>>(vtx, n, W10,b10,g10,e10, W11,b11,g11,e11,
                                   W12,b12,g12,e12, W2,b2,g2,e2, fiV, gV);
  k_front<<<256, 256, 0, stream>>>(pts, m, W10,b10,g10,e10, W11,b11,g11,e11,
                                   W12,b12,g12,e12, W2,b2,g2,e2, fiP, gP);
  k_back<<<(n + 7) / 8, 256, 0, stream>>>(fiV, gV, n, W30,b30,g30,e30,
                                          W31,b31,g31,e31, linW, linb, outV, hV);
  k_back<<<(m + 7) / 8, 256, 0, stream>>>(fiP, gP, m, W30,b30,g30,e30,
                                          W31,b31,g31,e31, linW, linb, outP, hP);
  k_sim<<<n / (16 * STRIPS), 256, 0, stream>>>(hV, hP, simv, idxv, m);
  k_corr<<<(n + 7) / 8, 256, 0, stream>>>(outV, outP, idxv, n,
                                          Wc, bc, gc, ec, corrW, corrb, outMask);
}
